// GNNModel_9337258901633
// MI455X (gfx1250) — compile-verified, matrix operand order checked by reading, not run on hardware
//
#include <hip/hip_runtime.h>

typedef __attribute__((ext_vector_type(16))) __bf16 bf16x16;
typedef __attribute__((ext_vector_type(8)))  float  f32x8;
typedef unsigned int u32x4 __attribute__((ext_vector_type(4)));
typedef int          i32x8 __attribute__((ext_vector_type(8)));
typedef int          i32x4 __attribute__((ext_vector_type(4)));

#define HD 4
#define CCH 128
#define HC 512
#define GG 64
#define A_STRIDE 34   // 32 K-dwords + 2 pad dwords (TDM pad) -> spreads LDS banks

// ---------------- utility: order-preserving float<->uint key for atomicMax ----
__device__ __forceinline__ unsigned fkey(float f) {
  unsigned b = __float_as_uint(f);
  return (b & 0x80000000u) ? ~b : (b | 0x80000000u);
}
__device__ __forceinline__ float funkey(unsigned k) {
  unsigned b = (k & 0x80000000u) ? (k & 0x7FFFFFFFu) : ~k;
  return __uint_as_float(b);
}

// ---------------- TDM 2D tile load: global (fp32, row-major) -> LDS -----------
// Descriptor per CDNA5 ISA 8.3/8.4: group0 = {flags, lds_addr, global_addr, type},
// group1 = {data_size/pad, tensor dims == tile dims, tile dims, row stride}.
// tile_w/tile_h in fp32 elements; row_stride = leading dimension (elements).
__device__ __forceinline__ void tdm_load_2d(unsigned lds_addr, const void* gaddr,
                                            unsigned tile_w, unsigned tile_h,
                                            unsigned long long row_stride,
                                            unsigned pad_en, unsigned pad_icode,
                                            unsigned pad_acode) {
#if defined(__gfx1250__) && __has_builtin(__builtin_amdgcn_tensor_load_to_lds)
  unsigned long long ga = (unsigned long long)gaddr;
  u32x4 g0;
  g0[0] = 1u;                                              // count=1, user mode
  g0[1] = lds_addr;                                        // LDS byte address
  g0[2] = (unsigned)(ga & 0xFFFFFFFFu);                    // global_addr[31:0]
  g0[3] = (unsigned)((ga >> 32) & 0x1FFFFFFu) | (2u << 30);// addr[56:32] | type=2
  i32x8 g1;
  g1[0] = (int)((2u << 16) | (pad_en << 20) | (pad_icode << 22) | (pad_acode << 25));
  g1[1] = (int)((tile_w & 0xFFFFu) << 16);                 // tensor_dim0 lo
  g1[2] = (int)(((tile_w >> 16) & 0xFFFFu) | ((tile_h & 0xFFFFu) << 16));
  g1[3] = (int)(((tile_h >> 16) & 0xFFFFu) | ((tile_w & 0xFFFFu) << 16)); // tile_dim0
  g1[4] = (int)(tile_h & 0xFFFFu);                         // tile_dim1, tile_dim2=0
  g1[5] = (int)(row_stride & 0xFFFFFFFFull);               // dim0 stride lo
  g1[6] = (int)((row_stride >> 32) & 0xFFFFull);           // dim0 stride hi
  g1[7] = 0;
  i32x4 z4 = {0, 0, 0, 0};
  i32x8 z8 = {0, 0, 0, 0, 0, 0, 0, 0};
  __builtin_amdgcn_tensor_load_to_lds(g0, g1, z4, z4, z8, 0);  // 6-arg form
#else
  (void)lds_addr; (void)gaddr; (void)tile_w; (void)tile_h;
  (void)row_stride; (void)pad_en; (void)pad_icode; (void)pad_acode;
#endif
}

// ---------------- bf16 WMMA GEMM with TDM-staged LDS panels -------------------
// C[M,N] = act(A[M,K] @ B[K,N] + bias). Block = 8 waves; each wave one 16x16
// tile. Block tile = (16*mtB) x (16*ntB) with ntB = min(8, N/16), mtB = 8/ntB.
// All shapes here are exact multiples, so EXEC stays all-ones for WMMA.
__global__ void k_gemm_bf16(const float* __restrict__ A, const float* __restrict__ B,
                            const float* __restrict__ bias, float* __restrict__ C,
                            int M, int N, int K, int act) {
  __shared__ float lA[32 * A_STRIDE];  // up to 32 rows x 32 k (stride 34)
  __shared__ float lB[32 * 128];       // 32 k x up to 128 cols

  const int lane   = threadIdx.x & 31;
  const int wave   = threadIdx.x >> 5;
  const int ntiles = N >> 4;
  const int ntB    = ntiles < 8 ? ntiles : 8;
  const int mtB    = 8 / ntB;
  const int mBlocks = (M + 16 * mtB - 1) / (16 * mtB);
  const int bm = blockIdx.x % mBlocks;
  const int bn = blockIdx.x / mBlocks;
  const int lmt = wave / ntB;
  const int lnt = wave % ntB;
  const int rowBase = bm * mtB * 16;             // block's first A row
  const int colBase = bn * ntB * 16;             // block's first B col
  if (rowBase >= M) return;                      // uniform over block

  const int half = lane >> 4;
  const int r    = lane & 15;
  const int bRow = ntB * 16;                     // LDS B row stride (fp32)

  f32x8 acc = {};
  for (int k0 = 0; k0 < K; k0 += 32) {
    __syncthreads();                             // LDS reuse fence
#if defined(__gfx1250__) && __has_builtin(__builtin_amdgcn_tensor_load_to_lds)
    if (threadIdx.x < 32) {                      // wave 0 drives the TDM
      // A panel: (mtB*16) rows x 32 k, pad 2 dwords per 32-dword row -> stride 34
      tdm_load_2d((unsigned)(unsigned long long)(void*)lA,
                  A + (size_t)rowBase * K + k0,
                  32u, (unsigned)(mtB * 16), (unsigned long long)K,
                  1u, 4u /*32 DW*/, 1u /*2 DW*/);
      // B panel: 32 k rows x (ntB*16) cols, contiguous rows in LDS
      tdm_load_2d((unsigned)(unsigned long long)(void*)lB,
                  B + (size_t)k0 * N + colBase,
                  (unsigned)bRow, 32u, (unsigned long long)N, 0u, 0u, 0u);
      __builtin_amdgcn_s_wait_tensorcnt(0);
    }
#else
    // Fallback: cooperative staging through VGPRs (float2, 8B aligned)
    for (int c = threadIdx.x; c < mtB * 16 * 16; c += 256) {
      int row = c >> 4, kc = (c & 15) << 1;
      *(float2*)&lA[row * A_STRIDE + kc] =
          *(const float2*)&A[(size_t)(rowBase + row) * K + k0 + kc];
    }
    for (int c = threadIdx.x; c < 32 * (bRow >> 1); c += 256) {
      int row = c / (bRow >> 1), cc = (c % (bRow >> 1)) << 1;
      *(float2*)&lB[row * bRow + cc] =
          *(const float2*)&B[(size_t)(k0 + row) * N + colBase + cc];
    }
#endif
    __syncthreads();

    // Fragment assembly per ISA 7.12.2 (16-bit A 16x32 striping; B mirrored)
    bf16x16 a, b;
#pragma unroll
    for (int j = 0; j < 16; ++j) {
      const int v  = j >> 1;
      const int kk = ((v & 4) ? 16 : 0) + (half ? 8 : 0) + ((v & 3) << 1) + (j & 1);
      a[j] = (__bf16)lA[(lmt * 16 + r) * A_STRIDE + kk];
      b[j] = (__bf16)lB[kk * bRow + lnt * 16 + r];
    }
    acc = __builtin_amdgcn_wmma_f32_16x16x32_bf16(false, a, false, b,
                                                  (short)0, acc, false, false);
  }

  const long long col  = colBase + lnt * 16 + r;
  const long long row0 = rowBase + lmt * 16 + (half ? 8 : 0);
  const float bv = bias ? bias[col] : 0.f;
#pragma unroll
  for (int i = 0; i < 8; ++i) {
    float v = acc[i] + bv;
    if (act) v = fmaxf(v, 0.f);
    C[(row0 + i) * N + col] = v;
  }
}

// ---------------- attention coefficients: a_s/a_d [N,H] -----------------------
__global__ void k_attn(const float* __restrict__ h, const float* __restrict__ att_src,
                       const float* __restrict__ att_dst, float* __restrict__ a_s,
                       float* __restrict__ a_d, int n) {
  long long tid = (long long)blockIdx.x * blockDim.x + threadIdx.x;
  int wid = (int)(tid >> 5);              // one wave per (node, head)
  int lane = threadIdx.x & 31;
  if (wid >= n * HD) return;
  int node = wid >> 2, head = wid & 3;
  const float* hp = h + (long long)node * HC + head * CCH;
  const float* as = att_src + head * CCH;
  const float* ad = att_dst + head * CCH;
  float ss = 0.f, sd = 0.f;
#pragma unroll
  for (int c = lane; c < CCH; c += 32) {
    float v = hp[c];
    ss += v * as[c];
    sd += v * ad[c];
  }
#pragma unroll
  for (int o = 16; o > 0; o >>= 1) {
    ss += __shfl_down(ss, o, 32);
    sd += __shfl_down(sd, o, 32);
  }
  if (lane == 0) { a_s[wid] = ss; a_d[wid] = sd; }
}

// ---------------- edge pass 1: segment max of leaky_relu attention ------------
__global__ void k_edge_max(const int* __restrict__ ei, const float* __restrict__ a_s,
                           const float* __restrict__ a_d, unsigned* __restrict__ mkey,
                           int E, int E1) {
  long long idx = (long long)blockIdx.x * blockDim.x + threadIdx.x;
  if (idx >= (long long)E1 * HD) return;
  int i = (int)(idx >> 2), hd = (int)(idx & 3);
  int s, d;
  if (i < E) { s = ei[i]; d = ei[E + i]; } else { s = d = i - E; }
  float e = a_s[s * HD + hd] + a_d[d * HD + hd];
  e = (e > 0.f) ? e : 0.2f * e;
  atomicMax(mkey + (long long)d * HD + hd, fkey(e));
}

// ---------------- edge pass 2: ee = exp(e - m[dst]); denom += ee --------------
__global__ void k_edge_expsum(const int* __restrict__ ei, const float* __restrict__ a_s,
                              const float* __restrict__ a_d, const unsigned* __restrict__ mkey,
                              float* __restrict__ denom, float* __restrict__ ee,
                              int E, int E1) {
  long long idx = (long long)blockIdx.x * blockDim.x + threadIdx.x;
  if (idx >= (long long)E1 * HD) return;
  int i = (int)(idx >> 2), hd = (int)(idx & 3);
  int s, d;
  if (i < E) { s = ei[i]; d = ei[E + i]; } else { s = d = i - E; }
  float e = a_s[s * HD + hd] + a_d[d * HD + hd];
  e = (e > 0.f) ? e : 0.2f * e;
  float m = funkey(mkey[(long long)d * HD + hd]);
  float v = __expf(e - m);
  ee[idx] = v;
  atomicAdd(denom + (long long)d * HD + hd, v);
}

// ---------------- edge pass 3: gat_out[dst] += h[src] * alpha (heavy) ---------
__global__ void k_scatter_gat(const int* __restrict__ ei, const float* __restrict__ h,
                              const float* __restrict__ ee, const float* __restrict__ denom,
                              float* __restrict__ gat, int E, int E1) {
  long long idx = (long long)blockIdx.x * blockDim.x + threadIdx.x;
  if (idx >= (long long)E1 * HC) return;
  int i = (int)(idx >> 9);
  int c = (int)(idx & (HC - 1));
  int hd = c >> 7;
  int s, d;
  if (i < E) { s = ei[i]; d = ei[E + i]; } else { s = d = i - E; }
  float alpha = ee[(long long)i * HD + hd] / denom[(long long)d * HD + hd];
  atomicAdd(gat + (long long)d * HC + c, h[(long long)s * HC + c] * alpha);
}

// ---------------- x1 = relu(gat + bias); g := x1 (GIN self term, eps=0) -------
__global__ void k_bias_relu_copy(float* __restrict__ gat, const float* __restrict__ bias,
                                 float* __restrict__ g, long long n) {
  long long idx = (long long)blockIdx.x * blockDim.x + threadIdx.x;
  if (idx >= n) return;
  float v = gat[idx] + bias[idx & (HC - 1)];
  v = fmaxf(v, 0.f);
  gat[idx] = v;   // becomes x1
  g[idx]   = v;   // GIN: (1+eps)*x1 with eps=0
}

// ---------------- GIN neighbor sum: g[dst] += x1[src] -------------------------
__global__ void k_scatter_gin(const int* __restrict__ ei, const float* __restrict__ x1,
                              float* __restrict__ g, int E) {
  long long idx = (long long)blockIdx.x * blockDim.x + threadIdx.x;
  if (idx >= (long long)E * HC) return;
  int i = (int)(idx >> 9);
  int c = (int)(idx & (HC - 1));
  int s = ei[i], d = ei[E + i];
  atomicAdd(g + (long long)d * HC + c, x1[(long long)s * HC + c]);
}

// ---------------- mean pool accumulation --------------------------------------
__global__ void k_pool(const float* __restrict__ h2, const int* __restrict__ batch,
                       float* __restrict__ pooled, float* __restrict__ cnt, int n) {
  long long idx = (long long)blockIdx.x * blockDim.x + threadIdx.x;
  if (idx >= (long long)n * 64) return;
  int node = (int)(idx >> 6);
  int c = (int)(idx & 63);
  int b = batch[node];
  atomicAdd(pooled + b * 64 + c, h2[idx]);
  if (c == 0) atomicAdd(cnt + b, 1.0f);
}

// ---------------- final tiny FC: out = (pooled/cnt) @ Wfc + bfc ---------------
__global__ void k_fc(const float* __restrict__ pooled, const float* __restrict__ cnt,
                     const float* __restrict__ Wfc, const float* __restrict__ bfc,
                     float* __restrict__ out) {
  int g = blockIdx.x;
  int j = threadIdx.x;
  float cn = fmaxf(cnt[g], 1.0f);
  float s = 0.f;
#pragma unroll
  for (int k = 0; k < 64; ++k)
    s += (pooled[g * 64 + k] / cn) * Wfc[k * 64 + j];
  out[g * 64 + j] = s + bfc[j];
}

// ------------------------------------------------------------------------------
extern "C" void kernel_launch(void* const* d_in, const int* in_sizes, int n_in,
                              void* d_out, int out_size, void* d_ws, size_t ws_size,
                              hipStream_t stream) {
  const float* x        = (const float*)d_in[0];
  const int*   ei       = (const int*)d_in[1];
  const int*   batch    = (const int*)d_in[2];
  const float* W_gat    = (const float*)d_in[3];
  const float* att_src  = (const float*)d_in[4];
  const float* att_dst  = (const float*)d_in[5];
  const float* bias_gat = (const float*)d_in[6];
  const float* W1       = (const float*)d_in[7];
  const float* b1       = (const float*)d_in[8];
  const float* W2       = (const float*)d_in[9];
  const float* b2       = (const float*)d_in[10];
  const float* Wfc      = (const float*)d_in[11];
  const float* bfc      = (const float*)d_in[12];
  float* out = (float*)d_out;

  const int Nn = in_sizes[0] / 32;   // 20000
  const int E  = in_sizes[1] / 2;    // 320000
  const int E1 = E + Nn;
  const int K1 = in_sizes[3] / HC;   // 32

  // ---- workspace carving (all fp32, 256B aligned) ----
  char* ws = (char*)d_ws;
  size_t off = 0;
  auto carve = [&](size_t bytes) -> void* {
    void* p = ws + off;
    off += (bytes + 255) & ~(size_t)255;
    return p;
  };
  float*    h      = (float*)carve((size_t)Nn * HC * 4);   // GAT features
  float*    gat    = (float*)carve((size_t)Nn * HC * 4);   // gat_out -> x1
  float*    g      = (float*)carve((size_t)Nn * HC * 4);   // x1 + neighbor sum
  float*    t      = (float*)carve((size_t)Nn * 128 * 4);  // relu(g@W1+b1)
  float*    h2     = (float*)carve((size_t)Nn * 64 * 4);   // t@W2+b2
  float*    a_s    = (float*)carve((size_t)Nn * HD * 4);
  float*    a_d    = (float*)carve((size_t)Nn * HD * 4);
  unsigned* mkey   = (unsigned*)carve((size_t)Nn * HD * 4);
  float*    denom  = (float*)carve((size_t)Nn * HD * 4);
  float*    ee     = (float*)carve((size_t)E1 * HD * 4);
  float*    pooled = (float*)carve((size_t)GG * 64 * 4);
  float*    cnt    = (float*)carve((size_t)GG * 4);

  auto nb = [](long long n, int bs) { return (unsigned)((n + bs - 1) / bs); };
  auto gemm_blocks = [](int M, int N) {
    int ntiles = N >> 4;
    int ntB = ntiles < 8 ? ntiles : 8;
    int mtB = 8 / ntB;
    return (unsigned)(((M + 16 * mtB - 1) / (16 * mtB)) * (ntiles / ntB));
  };

  // ---- zero accumulators (mkey=0 is below every real attention key) ----
  (void)hipMemsetAsync(gat,    0, (size_t)Nn * HC * 4, stream);
  (void)hipMemsetAsync(mkey,   0, (size_t)Nn * HD * 4, stream);
  (void)hipMemsetAsync(denom,  0, (size_t)Nn * HD * 4, stream);
  (void)hipMemsetAsync(pooled, 0, (size_t)GG * 64 * 4, stream);
  (void)hipMemsetAsync(cnt,    0, (size_t)GG * 4, stream);

  // 1) h = x @ W_gat        [Nn,32] @ [32,512]
  k_gemm_bf16<<<gemm_blocks(Nn, HC), 256, 0, stream>>>(x, W_gat, nullptr, h, Nn, HC, K1, 0);
  // 2) attention coefficients
  k_attn<<<nb((long long)Nn * HD * 32, 256), 256, 0, stream>>>(h, att_src, att_dst, a_s, a_d, Nn);
  // 3) segment softmax over dst (3 edge passes)
  k_edge_max   <<<nb((long long)E1 * HD, 256), 256, 0, stream>>>(ei, a_s, a_d, mkey, E, E1);
  k_edge_expsum<<<nb((long long)E1 * HD, 256), 256, 0, stream>>>(ei, a_s, a_d, mkey, denom, ee, E, E1);
  k_scatter_gat<<<nb((long long)E1 * HC, 256), 256, 0, stream>>>(ei, h, ee, denom, gat, E, E1);
  // 4) x1 = relu(gat_out + bias); g = x1
  k_bias_relu_copy<<<nb((long long)Nn * HC, 256), 256, 0, stream>>>(gat, bias_gat, g, (long long)Nn * HC);
  // 5) GIN: g += sum_{neighbors} x1
  k_scatter_gin<<<nb((long long)E * HC, 256), 256, 0, stream>>>(ei, gat, g, E);
  // 6) t = relu(g @ W1 + b1)   [Nn,512] @ [512,128]
  k_gemm_bf16<<<gemm_blocks(Nn, 128), 256, 0, stream>>>(g, W1, b1, t, Nn, 128, HC, 1);
  // 7) h2 = t @ W2 + b2        [Nn,128] @ [128,64]
  k_gemm_bf16<<<gemm_blocks(Nn, 64), 256, 0, stream>>>(t, W2, b2, h2, Nn, 64, 128, 0);
  // 8) mean pool + final FC
  k_pool<<<nb((long long)Nn * 64, 256), 256, 0, stream>>>(h2, batch, pooled, cnt, Nn);
  k_fc<<<GG, 64, 0, stream>>>(pooled, cnt, Wfc, bfc, out);
}
